// FlashscGPTLayer_21955872817239
// MI455X (gfx1250) — matrix operand order repaired and checked by reading, not compiled, not run on hardware
//
#include <hip/hip_runtime.h>
#include <hip/hip_bf16.h>

typedef __attribute__((ext_vector_type(16))) _Float16 v16h;
typedef __attribute__((ext_vector_type(8)))  _Float16 v8h;
typedef __attribute__((ext_vector_type(4)))  _Float16 v4h;
typedef __attribute__((ext_vector_type(8)))  float    v8f;

#define T_TOK 16384
#define H_DIM 1024
#define N_EXP 8
#define D_E   128
#define D_S   512
#define LDB   40   // padded LDS row stride (halves): 80B, 16B-aligned, conflict-free frag reads

// ---------------- WMMA fragment helpers (CDNA5 16x16x32 f16, wave32) ----------------
// A (16x32 f16): lane m=lane&15 is row; hi=lane>>4.  slot e -> K=k0+hi*16+e.
// One contiguous 32B run per lane.
__device__ __forceinline__ v16h load_a_row(const _Float16* rowk0, int hi) {
  const _Float16* p = rowk0 + hi * 16;
  v8h x0 = *(const v8h*)(p);
  v8h x1 = *(const v8h*)(p + 8);
  v16h a;
#pragma unroll
  for (int i = 0; i < 8; ++i) { a[i] = x0[i]; a[i + 8] = x1[i]; }
  return a;
}

// B (32x16 f16): lane n=lane&15 is column; hi=lane>>4. slot e -> K=k0+hi*16+e.
// Source is [N][K] row-major -> one contiguous 32B run of row n.
__device__ __forceinline__ v16h load_b_row(const _Float16* W, int ldk, int n, int k0, int hi) {
  const _Float16* p = W + (size_t)n * ldk + k0 + hi * 16;
  v8h x0 = *(const v8h*)(p);
  v8h x1 = *(const v8h*)(p + 8);
  v16h b;
#pragma unroll
  for (int i = 0; i < 8; ++i) { b[i] = x0[i]; b[i + 8] = x1[i]; }
  return b;
}

__device__ __forceinline__ v8f wmma16(const v16h& a, const v16h& b, const v8f& c) {
  return __builtin_amdgcn_wmma_f32_16x16x32_f16(false, a, false, b, (short)0, c, false, false);
}

// Cooperative stage of a 128(N) x 32(K) B panel into LDS (padded rows), 32B per thread.
__device__ __forceinline__ void stage_b_lds(_Float16* Bs, const _Float16* W, int ldk,
                                            int n0, int k0, int tid) {
  const int n  = tid >> 1;
  const int kp = (tid & 1) << 4;
  const _Float16* src = W + (size_t)(n0 + n) * ldk + k0 + kp;
  v8h a0 = *(const v8h*)src;
  v8h a1 = *(const v8h*)(src + 8);
  _Float16* d = Bs + n * LDB + kp;
  *(v8h*)d = a0;
  *(v8h*)(d + 8) = a1;
}

__device__ __forceinline__ v16h load_b_lds(const _Float16* Bs, int n, int hi) {
  const _Float16* p = Bs + n * LDB + hi * 16;
  v8h x0 = *(const v8h*)(p);
  v8h x1 = *(const v8h*)(p + 8);
  v16h b;
#pragma unroll
  for (int i = 0; i < 8; ++i) { b[i] = x0[i]; b[i + 8] = x1[i]; }
  return b;
}

// ---------------- small utility kernels ----------------
__global__ void zero_cnt_kernel(int* cnt) {
  if (threadIdx.x < N_EXP) cnt[threadIdx.x] = 0;
}

__global__ void cast_f32_f16_kernel(const float* __restrict__ src, _Float16* __restrict__ dst, int n) {
  int i = (blockIdx.x * 256 + threadIdx.x) * 4;
  if (i + 3 < n) {
    float4 v = *(const float4*)(src + i);
    v4h o;
    o[0] = (_Float16)v.x; o[1] = (_Float16)v.y; o[2] = (_Float16)v.z; o[3] = (_Float16)v.w;
    *(v4h*)(dst + i) = o;
  }
}

// ---------------- gate: softmax top-2 + per-expert compaction ----------------
__global__ void gate_kernel(const float* __restrict__ x, const float* __restrict__ Wg,
                            int* __restrict__ cnt, int* __restrict__ toks, float* __restrict__ wlist) {
  __shared__ float wgT[H_DIM * N_EXP];  // [h][e], 32KB
  for (int i = threadIdx.x; i < H_DIM * N_EXP; i += 256)
    wgT[i] = Wg[(i & 7) * H_DIM + (i >> 3)];
  __syncthreads();

  const int t = blockIdx.x * 256 + threadIdx.x;
  const float* xr = x + (size_t)t * H_DIM;
  float acc[N_EXP];
#pragma unroll
  for (int e = 0; e < N_EXP; ++e) acc[e] = 0.f;
  for (int h = 0; h < H_DIM; h += 4) {
    float4 xv = *(const float4*)(xr + h);
#pragma unroll
    for (int e = 0; e < N_EXP; ++e) {
      acc[e] += xv.x * wgT[(h + 0) * N_EXP + e];
      acc[e] += xv.y * wgT[(h + 1) * N_EXP + e];
      acc[e] += xv.z * wgT[(h + 2) * N_EXP + e];
      acc[e] += xv.w * wgT[(h + 3) * N_EXP + e];
    }
  }
  // exact top-2 (ties keep lowest index, matching top_k)
  int i1 = 0;
#pragma unroll
  for (int e = 1; e < N_EXP; ++e) if (acc[e] > acc[i1]) i1 = e;
  int i2 = (i1 == 0) ? 1 : 0;
#pragma unroll
  for (int e = 0; e < N_EXP; ++e) if (e != i1 && acc[e] > acc[i2]) i2 = e;
  // renormalized softmax weights: denominator cancels -> logistic of logit gap
  float g  = __expf(acc[i2] - acc[i1]);
  float w1 = 1.f / (1.f + g);
  float w2 = g * w1;
  int p1 = atomicAdd(&cnt[i1], 1);
  toks[i1 * T_TOK + p1] = t; wlist[i1 * T_TOK + p1] = w1;
  int p2 = atomicAdd(&cnt[i2], 1);
  toks[i2 * T_TOK + p2] = t; wlist[i2 * T_TOK + p2] = w2;
}

// ---------------- shared expert fc1: Hs = relu(X @ Ws1^T + bs1), f16 out ----------------
// block tile 128 tok x 128 N, wave tile 16 x 128, B panel double-buffered in LDS.
__global__ void shared_fc1_kernel(const _Float16* __restrict__ Xh, const _Float16* __restrict__ Ws1h,
                                  const float* __restrict__ bs1, _Float16* __restrict__ Hs) {
  __shared__ _Float16 Bs[2][128 * LDB];  // 20KB
  const int tid = threadIdx.x;
  const int wave = tid >> 5, lane = tid & 31;
  const int hi = lane >> 4, ln = lane & 15;
  const int m0 = blockIdx.x * 128 + wave * 16;
  const int n0 = blockIdx.y * 128;
  const _Float16* arow = Xh + (size_t)(m0 + ln) * H_DIM;

  v8f acc[8];
#pragma unroll
  for (int j = 0; j < 8; ++j) acc[j] = (v8f)0.0f;

  stage_b_lds(Bs[0], Ws1h, H_DIM, n0, 0, tid);
  __syncthreads();
  for (int k0 = 0; k0 < H_DIM; k0 += 32) {
    const int cur = (k0 >> 5) & 1;
    if (k0 + 32 < H_DIM) stage_b_lds(Bs[cur ^ 1], Ws1h, H_DIM, n0, k0 + 32, tid);
    v16h a = load_a_row(arow + k0, hi);
#pragma unroll
    for (int j = 0; j < 8; ++j) {
      v16h b = load_b_lds(Bs[cur], 16 * j + ln, hi);
      acc[j] = wmma16(a, b, acc[j]);
    }
    __syncthreads();
  }
#pragma unroll
  for (int j = 0; j < 8; ++j) {
    const int ncol = n0 + 16 * j + ln;
    const float bias = bs1[ncol];
#pragma unroll
    for (int r = 0; r < 8; ++r) {
      float v = acc[j][r] + bias;
      v = v > 0.f ? v : 0.f;
      Hs[(size_t)(m0 + hi * 8 + r) * D_S + ncol] = (_Float16)v;
    }
  }
}

// ---------------- shared expert fc2: out = Hs @ Ws2^T + bs2 (initializes out) ----------------
__global__ void shared_fc2_kernel(const _Float16* __restrict__ Hs, const _Float16* __restrict__ Ws2h,
                                  const float* __restrict__ bs2, float* __restrict__ out) {
  __shared__ _Float16 Bs[2][128 * LDB];
  const int tid = threadIdx.x;
  const int wave = tid >> 5, lane = tid & 31;
  const int hi = lane >> 4, ln = lane & 15;
  const int m0 = blockIdx.x * 128 + wave * 16;
  const int n0 = blockIdx.y * 128;
  const _Float16* arow = Hs + (size_t)(m0 + ln) * D_S;

  v8f acc[8];
#pragma unroll
  for (int j = 0; j < 8; ++j) acc[j] = (v8f)0.0f;

  stage_b_lds(Bs[0], Ws2h, D_S, n0, 0, tid);
  __syncthreads();
  for (int k0 = 0; k0 < D_S; k0 += 32) {
    const int cur = (k0 >> 5) & 1;
    if (k0 + 32 < D_S) stage_b_lds(Bs[cur ^ 1], Ws2h, D_S, n0, k0 + 32, tid);
    v16h a = load_a_row(arow + k0, hi);
#pragma unroll
    for (int j = 0; j < 8; ++j) {
      v16h b = load_b_lds(Bs[cur], 16 * j + ln, hi);
      acc[j] = wmma16(a, b, acc[j]);
    }
    __syncthreads();
  }
#pragma unroll
  for (int j = 0; j < 8; ++j) {
    const int ncol = n0 + 16 * j + ln;
    const float bias = bs2[ncol];
#pragma unroll
    for (int r = 0; r < 8; ++r)
      out[(size_t)(m0 + hi * 8 + r) * H_DIM + ncol] = acc[j][r] + bias;
  }
}

// ---------------- routed experts: gather -> fc1 -> relu*w (LDS) -> fc2 -> atomic scatter ----------------
__global__ void routed_expert_kernel(const _Float16* __restrict__ Xh,
                                     const _Float16* __restrict__ W1h, const float* __restrict__ b1,
                                     const _Float16* __restrict__ W2h, const float* __restrict__ b2,
                                     const int* __restrict__ cnt, const int* __restrict__ toks,
                                     const float* __restrict__ wlist, float* __restrict__ out) {
  __shared__ _Float16 h1[128 * D_E];     // 32KB gate-weighted fc1 activations
  __shared__ _Float16 Bs[2][128 * LDB];  // 20KB W1 panel
  __shared__ int   s_tok[128];
  __shared__ float s_w[128];

  const int e = blockIdx.y;
  const int nAssigned = cnt[e];
  const int base = blockIdx.x * 128;
  if (base >= nAssigned) return;  // uniform across block; before any barrier

  const int tid = threadIdx.x;
  if (tid < 128) {
    int s = base + tid;
    int t = 0; float w = 0.f;
    if (s < nAssigned) { t = toks[e * T_TOK + s]; w = wlist[e * T_TOK + s]; }
    s_tok[tid] = t; s_w[tid] = w;
  }
  __syncthreads();

  const int wave = tid >> 5, lane = tid & 31;
  const int hi = lane >> 4, ln = lane & 15;
  const int m0 = wave * 16;

  // ---- fc1: this wave's 16 gathered tokens x all 128 DE cols (B staged in LDS) ----
  {
    const _Float16* arow = Xh + (size_t)s_tok[m0 + ln] * H_DIM;
    const _Float16* We = W1h + (size_t)e * D_E * H_DIM;
    v8f acc[8];
#pragma unroll
    for (int j = 0; j < 8; ++j) acc[j] = (v8f)0.0f;

    stage_b_lds(Bs[0], We, H_DIM, 0, 0, tid);
    __syncthreads();
    for (int k0 = 0; k0 < H_DIM; k0 += 32) {
      const int cur = (k0 >> 5) & 1;
      if (k0 + 32 < H_DIM) stage_b_lds(Bs[cur ^ 1], We, H_DIM, 0, k0 + 32, tid);
      v16h a = load_a_row(arow + k0, hi);
#pragma unroll
      for (int j = 0; j < 8; ++j) {
        v16h b = load_b_lds(Bs[cur], 16 * j + ln, hi);
        acc[j] = wmma16(a, b, acc[j]);
      }
      __syncthreads();
    }
    const float* b1e = b1 + e * D_E;
#pragma unroll
    for (int j = 0; j < 8; ++j) {
      const float bias = b1e[16 * j + ln];
#pragma unroll
      for (int r = 0; r < 8; ++r) {
        const int m = m0 + hi * 8 + r;
        float v = acc[j][r] + bias;
        v = (v > 0.f ? v : 0.f) * s_w[m];  // fold gate weight (fc2 is linear)
        h1[m * D_E + 16 * j + ln] = (_Float16)v;
      }
    }
  }
  __syncthreads();

  // ---- fc2: 16 tokens x H=1024 in 128-col chunks; A from LDS h1 ----
  {
    const _Float16* arow = &h1[(m0 + ln) * D_E];
    const _Float16* We = W2h + (size_t)e * H_DIM * D_E;
    const float* b2e = b2 + e * H_DIM;
    for (int nc = 0; nc < H_DIM; nc += 128) {
      v8f acc[8];
#pragma unroll
      for (int j = 0; j < 8; ++j) acc[j] = (v8f)0.0f;
#pragma unroll
      for (int k0 = 0; k0 < D_E; k0 += 32) {
        v16h a = load_a_row(arow + k0, hi);
#pragma unroll
        for (int j = 0; j < 8; ++j) {
          v16h b = load_b_row(We, D_E, nc + 16 * j + ln, k0, hi);
          acc[j] = wmma16(a, b, acc[j]);
        }
      }
#pragma unroll
      for (int j = 0; j < 8; ++j) {
        const int ncol = nc + 16 * j + ln;
        const float bias = b2e[ncol];
#pragma unroll
        for (int r = 0; r < 8; ++r) {
          const int m = m0 + hi * 8 + r;
          const float w = s_w[m];
          if (w != 0.f)  // skips padded slots; reconverges before next WMMA chunk
            atomicAdd(out + (size_t)s_tok[m] * H_DIM + ncol, acc[j][r] + w * bias);
        }
      }
    }
  }
}

// ---------------- host ----------------
extern "C" void kernel_launch(void* const* d_in, const int* in_sizes, int n_in,
                              void* d_out, int out_size, void* d_ws, size_t ws_size,
                              hipStream_t stream) {
  (void)in_sizes; (void)n_in; (void)out_size; (void)ws_size;
  const float* x   = (const float*)d_in[0];
  const float* Wg  = (const float*)d_in[1];
  const float* W1  = (const float*)d_in[2];
  const float* b1  = (const float*)d_in[3];
  const float* W2  = (const float*)d_in[4];
  const float* b2  = (const float*)d_in[5];
  const float* Ws1 = (const float*)d_in[6];
  const float* bs1 = (const float*)d_in[7];
  const float* Ws2 = (const float*)d_in[8];
  const float* bs2 = (const float*)d_in[9];
  float* out = (float*)d_out;

  char* ws = (char*)d_ws;
  size_t cur = 0;
  auto alloc = [&](size_t bytes) -> void* {
    void* p = ws + cur;
    cur = (cur + bytes + 255) & ~(size_t)255;
    return p;
  };
  _Float16* Xh   = (_Float16*)alloc((size_t)T_TOK * H_DIM * 2);
  _Float16* W1h  = (_Float16*)alloc((size_t)N_EXP * D_E * H_DIM * 2);
  _Float16* W2h  = (_Float16*)alloc((size_t)N_EXP * H_DIM * D_E * 2);
  _Float16* Ws1h = (_Float16*)alloc((size_t)D_S * H_DIM * 2);
  _Float16* Ws2h = (_Float16*)alloc((size_t)H_DIM * D_S * 2);
  _Float16* Hs   = (_Float16*)alloc((size_t)T_TOK * D_S * 2);
  int*      cnt  = (int*)alloc(N_EXP * sizeof(int));
  int*      toks = (int*)alloc((size_t)N_EXP * T_TOK * sizeof(int));
  float*    wlst = (float*)alloc((size_t)N_EXP * T_TOK * sizeof(float));

  zero_cnt_kernel<<<1, 32, 0, stream>>>(cnt);

  auto cast = [&](const float* s, _Float16* d, int n) {
    cast_f32_f16_kernel<<<(n / 4 + 255) / 256, 256, 0, stream>>>(s, d, n);
  };
  cast(x,   Xh,   T_TOK * H_DIM);
  cast(W1,  W1h,  N_EXP * D_E * H_DIM);
  cast(W2,  W2h,  N_EXP * H_DIM * D_E);
  cast(Ws1, Ws1h, D_S * H_DIM);
  cast(Ws2, Ws2h, H_DIM * D_S);

  gate_kernel<<<T_TOK / 256, 256, 0, stream>>>(x, Wg, cnt, toks, wlst);

  shared_fc1_kernel<<<dim3(T_TOK / 128, D_S / 128), 256, 0, stream>>>(Xh, Ws1h, bs1, Hs);
  shared_fc2_kernel<<<dim3(T_TOK / 128, H_DIM / 128), 256, 0, stream>>>(Hs, Ws2h, bs2, out);

  routed_expert_kernel<<<dim3(T_TOK / 128, N_EXP), 256, 0, stream>>>(
      Xh, W1h, b1, W2h, b2, cnt, toks, wlst, out);
}
